// StyleMultiHeadAttention_52235392254455
// MI455X (gfx1250) — matrix operand order repaired and checked by reading, not compiled, not run on hardware
//
#include <hip/hip_runtime.h>

// Problem sizes (fixed by the reference).
#define B_ 64
#define L_ 1024
#define E_ 512
#define H_ 8
#define D_ 64
#define SCALE_ 0.125f   // D^-0.5

typedef __attribute__((ext_vector_type(16))) _Float16 v16h;
typedef __attribute__((ext_vector_type(8)))  float    v8f;
typedef __attribute__((ext_vector_type(4)))  float    f32x4;   // native vector
                                                               // (HIP float4 is a
                                                               // class; builtins
                                                               // reject it)

// ---------------------------------------------------------------------------
// Kernel 1: q[b,e] = style[b,:] . Wq[e,:] + bq[e]                (B x E, tiny)
// ---------------------------------------------------------------------------
__global__ void qproj_kernel(const float* __restrict__ style,
                             const float* __restrict__ Wq,
                             const float* __restrict__ bq,
                             float* __restrict__ q) {
  const int e = blockIdx.x * blockDim.x + threadIdx.x;
  const int b = blockIdx.y;
  const f32x4* s4 = (const f32x4*)(style + (size_t)b * E_);
  const f32x4* w4 = (const f32x4*)(Wq + (size_t)e * E_);
  float acc = 0.f;
#pragma unroll 4
  for (int j = 0; j < E_ / 4; ++j) {
    f32x4 x = s4[j], y = w4[j];
    acc += x.x * y.x + x.y * y.y + x.z * y.z + x.w * y.w;
  }
  q[(size_t)b * E_ + e] = acc + bq[e];
}

// ---------------------------------------------------------------------------
// Kernel 2: u[b,h,j] = sum_d q[b,h*D+d] * Wk[h*D+d, j]           (B x H x E)
// (bk cancels under softmax: uniform shift per (b,h))
// ---------------------------------------------------------------------------
__global__ void ukernel(const float* __restrict__ q,
                        const float* __restrict__ Wk,
                        float* __restrict__ u) {
  const int j = blockIdx.x * blockDim.x + threadIdx.x;
  const int h = blockIdx.y;
  const int b = blockIdx.z;
  const float* qh = q + (size_t)b * E_ + h * D_;
  float acc = 0.f;
#pragma unroll 8
  for (int d = 0; d < D_; ++d)
    acc += qh[d] * Wk[(size_t)(h * D_ + d) * E_ + j];   // coalesced over j
  u[((size_t)b * H_ + h) * E_ + j] = acc;
}

// ---------------------------------------------------------------------------
// Kernel 3 (WMMA): scores[b,h,l] = SCALE * text[b,l,:] . u[b,h,:]  (+mask)
// Per-batch GEMM: A = text tile (16 rows x 32 k, f16), B = u (32 k x 16 heads,
// heads 8..15 zero-padded). One wave per 16-row tile; 8 waves / block.
// ---------------------------------------------------------------------------
__global__ void scores_kernel(const float* __restrict__ text,
                              const float* __restrict__ u,
                              const int*   __restrict__ mask,
                              float* __restrict__ scores) {
  const int lane = threadIdx.x & 31;
  const int wave = threadIdx.x >> 5;
  const int b    = blockIdx.y;
  const int m0   = (blockIdx.x * 8 + wave) * 16;   // row tile start in L
  const int half = lane >> 4;                       // 0: lanes 0-15, 1: 16-31
  const int q15  = lane & 15;

  const float* trow  = text + ((size_t)b * L_ + m0 + q15) * E_;        // A row
  const float* urow  = u + ((size_t)b * H_ + (q15 & 7)) * E_;          // B col
  const float  bmask = (q15 < H_) ? 1.f : 0.f;                         // pad N

  v8f c = {};
  for (int k0 = 0; k0 < E_; k0 += 32) {
    __builtin_prefetch(trow + k0 + 64, 0, 0);       // global_prefetch_b8
    v16h a, bm;
    // A: contiguous k order (same as B). lane<16: K 0..15; lane>=16: K 16..31
    const float* pa = trow + k0 + half * 16;
#pragma unroll
    for (int i = 0; i < 8; ++i) {
      a[i]     = (_Float16)pa[i];
      a[i + 8] = (_Float16)pa[i + 8];
    }
    // B: contiguous k order (same as A). lanes 0-15 hold K 0..15, lanes 16-31 hold K 16..31
    const float* pb = urow + k0 + half * 16;
#pragma unroll
    for (int i = 0; i < 16; ++i) bm[i] = (_Float16)(pb[i] * bmask);

    c = __builtin_amdgcn_wmma_f32_16x16x32_f16(false, a, false, bm,
                                               (short)0, c, false, false);
  }
  // C layout: VGPR r -> M = r + 8*half, N = lane&15 (= head)
  if (q15 < H_) {
#pragma unroll
    for (int r = 0; r < 8; ++r) {
      const int l = m0 + half * 8 + r;
      float sc = c[r] * SCALE_;
      if (mask[(size_t)b * L_ + l] == 0) sc = -3.0e38f;
      scores[((size_t)b * H_ + q15) * L_ + l] = sc;
    }
  }
}

// ---------------------------------------------------------------------------
// Kernel 4: masked softmax over L per (b,h), in place.
// ---------------------------------------------------------------------------
__global__ void softmax_kernel(float* __restrict__ scores) {
  const int h = blockIdx.x, b = blockIdx.y, t = threadIdx.x;
  float* s = scores + ((size_t)b * H_ + h) * L_;
  __shared__ float red[256];

  float mx = -3.4e38f;
  for (int i = t; i < L_; i += 256) mx = fmaxf(mx, s[i]);
  red[t] = mx; __syncthreads();
  for (int w = 128; w > 0; w >>= 1) { if (t < w) red[t] = fmaxf(red[t], red[t + w]); __syncthreads(); }
  mx = red[0]; __syncthreads();

  float sum = 0.f;
  for (int i = t; i < L_; i += 256) { float p = __expf(s[i] - mx); s[i] = p; sum += p; }
  red[t] = sum; __syncthreads();
  for (int w = 128; w > 0; w >>= 1) { if (t < w) red[t] += red[t + w]; __syncthreads(); }
  const float inv = 1.f / red[0];
  __syncthreads();
  for (int i = t; i < L_; i += 256) s[i] *= inv;
}

// ---------------------------------------------------------------------------
// Kernel 5 (WMMA + async-LDS): t[b,h,e] = sum_l attn[b,h,l] * text[b,l,e]
// One workgroup (8 waves) per batch. For each 32-row chunk of text (64 KB,
// contiguous), all 256 threads stage it into LDS with
// GLOBAL_LOAD_ASYNC_TO_LDS_B128 (ASYNCcnt path), then each wave computes 4
// column tiles of 16, reusing one attn A-fragment across the 4 WMMAs.
// ---------------------------------------------------------------------------
__global__ void tproj_kernel(const float* __restrict__ text,
                             const float* __restrict__ attn,
                             float* __restrict__ tmat) {
  __shared__ float tile[32 * E_];                  // 64 KB staging buffer

  const int lane = threadIdx.x & 31;
  const int wave = threadIdx.x >> 5;
  const int b    = blockIdx.x;
  const int half = lane >> 4;
  const int q15  = lane & 15;

  const float* arow  = attn + ((size_t)b * H_ + (q15 & 7)) * L_;
  const float  amask = (q15 < H_) ? 1.f : 0.f;     // pad heads 8..15 -> 0

  const unsigned int lds_base =
      (unsigned int)(unsigned long long)(&tile[0]);       // low 32b = LDS offset
  const unsigned int my_off = (unsigned int)threadIdx.x * 16u;  // 16B per thread

  v8f acc0 = {}, acc1 = {}, acc2 = {}, acc3 = {};

  for (int k0 = 0; k0 < L_; k0 += 32) {
    // ---- stage text[b, k0..k0+31, :] (contiguous 64 KB) into LDS ----
    const unsigned long long gsrc =
        (unsigned long long)(const void*)(text + ((size_t)b * L_ + k0) * E_);
#pragma unroll
    for (int i = 0; i < 16; ++i) {
      const unsigned int off = my_off + (unsigned int)i * 4096u;  // 256 thr * 16B
      const unsigned int la  = lds_base + off;
      const unsigned long long ga = gsrc + off;
      asm volatile("global_load_async_to_lds_b128 %0, %1, off"
                   :: "v"(la), "v"(ga) : "memory");
    }
    asm volatile("s_wait_asynccnt 0" ::: "memory");
    __syncthreads();

    // ---- A fragment: attn rows (heads), K = l-chunk; shared by 4 tiles ----
    v16h a;
    const float* pa = arow + k0 + half * 16;
#pragma unroll
    for (int i = 0; i < 8; ++i) {
      a[i]     = (_Float16)(pa[i] * amask);
      a[i + 8] = (_Float16)(pa[i + 8] * amask);
    }

    // ---- 4 column tiles per wave, B fragments from LDS ----
#pragma unroll
    for (int tt = 0; tt < 4; ++tt) {
      const int n0 = (wave * 4 + tt) * 16;
      const float* lb = &tile[(half * 16) * E_ + n0 + q15];
      v16h bm;
#pragma unroll
      for (int i = 0; i < 16; ++i) bm[i] = (_Float16)lb[(size_t)i * E_];
      v8f* acc = (tt == 0) ? &acc0 : (tt == 1) ? &acc1 : (tt == 2) ? &acc2 : &acc3;
      *acc = __builtin_amdgcn_wmma_f32_16x16x32_f16(false, a, false, bm,
                                                    (short)0, *acc, false, false);
    }
    __syncthreads();   // protect LDS before next chunk's async writes
  }

  // VGPR r -> M = r + 8*half; only heads 0..7 (half==0) are real.
  if (half == 0) {
#pragma unroll
    for (int tt = 0; tt < 4; ++tt) {
      const int n0 = (wave * 4 + tt) * 16;
      const v8f* acc = (tt == 0) ? &acc0 : (tt == 1) ? &acc1 : (tt == 2) ? &acc2 : &acc3;
#pragma unroll
      for (int r = 0; r < 8; ++r)
        tmat[((size_t)b * H_ + r) * E_ + n0 + q15] = (*acc)[r];
    }
  }
}

// ---------------------------------------------------------------------------
// Kernel 6: orow[b,e] = Wv[e,:] . t[b, e/D, :] + bv[e]           (B x E, tiny)
// ---------------------------------------------------------------------------
__global__ void vproj_kernel(const float* __restrict__ Wv,
                             const float* __restrict__ bv,
                             const float* __restrict__ tmat,
                             float* __restrict__ orow) {
  const int e = blockIdx.x * blockDim.x + threadIdx.x;
  const int b = blockIdx.y;
  const int h = e / D_;
  const f32x4* w4 = (const f32x4*)(Wv + (size_t)e * E_);
  const f32x4* t4 = (const f32x4*)(tmat + ((size_t)b * H_ + h) * E_);
  float acc = 0.f;
#pragma unroll 4
  for (int j = 0; j < E_ / 4; ++j) {
    f32x4 x = w4[j], y = t4[j];
    acc += x.x * y.x + x.y * y.y + x.z * y.z + x.w * y.w;
  }
  orow[(size_t)b * E_ + e] = acc + bv[e];
}

// ---------------------------------------------------------------------------
// Kernel 7: broadcast out[b,l,:] = orow[b,:]  (134 MB stream, NT stores so the
// text working set stays resident in the 192 MB L2 between passes)
// ---------------------------------------------------------------------------
__global__ void bcast_kernel(const float* __restrict__ orow,
                             float* __restrict__ out) {
  const size_t idx = (size_t)blockIdx.x * blockDim.x + threadIdx.x; // f32x4 id
  const size_t e4 = idx % (E_ / 4);
  const size_t b  = (idx / (E_ / 4)) / L_;
  const f32x4 v = ((const f32x4*)orow)[b * (E_ / 4) + e4];
  __builtin_nontemporal_store(v, ((f32x4*)out) + idx);
}

// ---------------------------------------------------------------------------
extern "C" void kernel_launch(void* const* d_in, const int* in_sizes, int n_in,
                              void* d_out, int out_size, void* d_ws, size_t ws_size,
                              hipStream_t stream) {
  const float* style = (const float*)d_in[0];
  const float* text  = (const float*)d_in[1];
  const int*   mask  = (const int*)d_in[2];
  const float* Wq    = (const float*)d_in[3];
  const float* bq    = (const float*)d_in[4];
  const float* Wk    = (const float*)d_in[5];
  // d_in[6] = bk: uniform shift per (b,h) under softmax -> cancels, unused.
  const float* Wv    = (const float*)d_in[7];
  const float* bv    = (const float*)d_in[8];
  float* out = (float*)d_out;

  // Workspace layout (bytes): q 128K | u 1M | scores/attn 2M | t 1M | orow 128K
  char* ws = (char*)d_ws;
  float* q      = (float*)(ws);
  float* u      = (float*)(ws + (size_t)131072);
  float* scores = (float*)(ws + (size_t)131072 + 1048576);
  float* tmat   = (float*)(ws + (size_t)131072 + 1048576 + 2097152);
  float* orow   = (float*)(ws + (size_t)131072 + 1048576 + 2097152 + 1048576);

  qproj_kernel <<<dim3(E_ / 256, B_),        256, 0, stream>>>(style, Wq, bq, q);
  ukernel      <<<dim3(E_ / 256, H_, B_),    256, 0, stream>>>(q, Wk, u);
  scores_kernel<<<dim3(L_ / 128, B_),        256, 0, stream>>>(text, u, mask, scores);
  softmax_kernel<<<dim3(H_, B_),             256, 0, stream>>>(scores);
  tproj_kernel <<<dim3(B_),                  256, 0, stream>>>(text, scores, tmat);
  vproj_kernel <<<dim3(E_ / 256, B_),        256, 0, stream>>>(Wv, bv, tmat, orow);
  bcast_kernel <<<dim3((B_ * L_ * E_ / 4) / 256), 256, 0, stream>>>(orow, out);
}
